// KNN_VC_40029095199189
// MI455X (gfx1250) — matrix operand order repaired and checked by reading, not compiled, not run on hardware
//
#include <hip/hip_runtime.h>

typedef __attribute__((ext_vector_type(16))) _Float16 v16h;
typedef __attribute__((ext_vector_type(8)))  _Float16 v8h;
typedef __attribute__((ext_vector_type(8)))  float    v8f;

#define DDIM   1024
#define NQ     2048
#define NM     32768
#define TOPK   4
#define CHUNK  2048                    // columns of M handled per block
#define NCHUNKS (NM / CHUNK)           // 16
#define COLS_PER_ITER 256              // 8 waves * 32 cols (dual accumulators)
#define ITERS (CHUNK / COLS_PER_ITER)  // 8
#define QPAD  8                        // f16 pad: row stride 1032 halves = 2064B (16B aligned, 4-bank skew)

union Frag { v16h v; v8h h[2]; };

__device__ __forceinline__ v8f wmma_f16(v16h a, v16h b, v8f c) {
    return __builtin_amdgcn_wmma_f32_16x16x32_f16(false, a, false, b, (short)0, c,
                                                  false, false);
}

__device__ __forceinline__ void ins4(float v, int i, float bv[TOPK], int bi[TOPK]) {
    if (v > bv[TOPK - 1]) {
        bv[TOPK - 1] = v; bi[TOPK - 1] = i;
        #pragma unroll
        for (int t = TOPK - 1; t > 0; --t) {
            if (bv[t] > bv[t - 1]) {
                float tv = bv[t]; bv[t] = bv[t - 1]; bv[t - 1] = tv;
                int   ti = bi[t]; bi[t] = bi[t - 1]; bi[t - 1] = ti;
            }
        }
    }
}

// ---------------------------------------------------------------------------
// Kernel 1: L2-normalize each row of a [nrows, 1024] f32 matrix into f16.
// grid.x = nrows, 256 threads.
// ---------------------------------------------------------------------------
__global__ __launch_bounds__(256) void normalize_f16(const float* __restrict__ src,
                                                     _Float16* __restrict__ dst) {
    const int row = blockIdx.x;
    const float* s = src + (size_t)row * DDIM;
    float ss = 0.f;
    for (int i = threadIdx.x; i < DDIM; i += 256) {
        float v = s[i];
        ss += v * v;
    }
    __shared__ float red[256];
    red[threadIdx.x] = ss;
    __syncthreads();
    for (int off = 128; off > 0; off >>= 1) {
        if (threadIdx.x < off) red[threadIdx.x] += red[threadIdx.x + off];
        __syncthreads();
    }
    const float inv = rsqrtf(red[0] + 1e-20f);
    _Float16* d = dst + (size_t)row * DDIM;
    for (int i = threadIdx.x; i < DDIM; i += 256) {
        d[i] = (_Float16)(s[i] * inv);
    }
}

// ---------------------------------------------------------------------------
// Kernel 2: fused cosine-sim GEMM + per-row top-4 within a column chunk.
// grid = (NCHUNKS, NQ/16), 256 threads (8 waves).
// Each wave owns two 16x16 output tiles (32 columns) sharing one A fragment;
// the K loop is software-pipelined: fragments for k+32 are loaded while the
// WMMAs for k execute, so s_wait covers loads a full iteration old.
// ---------------------------------------------------------------------------
__global__ __launch_bounds__(256) void gemm_topk(const _Float16* __restrict__ Qn,
                                                 const _Float16* __restrict__ Mn,
                                                 float* __restrict__ cand_val,
                                                 int* __restrict__ cand_idx) {
    const int chunk   = blockIdx.x;
    const int rowBase = blockIdx.y * 16;
    const int colBase = chunk * CHUNK;

    __shared__ _Float16 qs[16][DDIM + QPAD];        // ~32.9 KB, 16B-aligned rows
    __shared__ float    sc[16][COLS_PER_ITER];      // 16 KB score tile
    __shared__ float    mval[16][16 * TOPK];        // 4 KB merge buffer
    __shared__ int      midx[16][16 * TOPK];        // 4 KB

    // Stage the 16x1024 query tile into LDS once.
    for (int i = threadIdx.x; i < 16 * DDIM; i += 256) {
        int r = i >> 10, c = i & (DDIM - 1);
        qs[r][c] = Qn[(size_t)(rowBase + r) * DDIM + c];
    }
    __syncthreads();

    const int lane = threadIdx.x & 31;
    const int wave = threadIdx.x >> 5;      // 0..7
    const int nsub = lane & 15;             // column / A-row within tile
    const int hh   = lane >> 4;             // half-wave

    // Selection ownership: thread = (row 0..15, group 0..15), 16 cols per group.
    const int selRow = threadIdx.x >> 4;
    const int selGrp = threadIdx.x & 15;

    float bestV[TOPK] = {-2e30f, -2e30f, -2e30f, -2e30f};
    int   bestI[TOPK] = {0, 0, 0, 0};

    for (int it = 0; it < ITERS; ++it) {
        const int colT = colBase + it * COLS_PER_ITER + wave * 32;
        // B fragment sources: rows (colT+nsub) and (colT+16+nsub) of Mn,
        // K runs of 8 halves at 8*hh and 16+8*hh (matches A table order).
        const _Float16* brow0 = Mn + (size_t)(colT + nsub) * DDIM + hh * 8;
        const _Float16* brow1 = brow0 + (size_t)16 * DDIM;

        // Prefetch next iteration's B rows into cache (global_prefetch_b8).
        if (it + 1 < ITERS) {
            __builtin_prefetch(brow0 + (size_t)COLS_PER_ITER * DDIM, 0, 0);
            __builtin_prefetch(brow1 + (size_t)COLS_PER_ITER * DDIM, 0, 0);
        }

        // ---- software-pipelined K loop ----
        Frag aC, b0C, b1C;
        aC.h[0]  = *(const v8h*)&qs[nsub][8 * hh];
        aC.h[1]  = *(const v8h*)&qs[nsub][16 + 8 * hh];
        b0C.h[0] = *(const v8h*)(brow0);
        b0C.h[1] = *(const v8h*)(brow0 + 16);
        b1C.h[0] = *(const v8h*)(brow1);
        b1C.h[1] = *(const v8h*)(brow1 + 16);

        v8f acc0 = {}, acc1 = {};
        #pragma unroll 4
        for (int k = 32; k < DDIM; k += 32) {
            Frag aN, b0N, b1N;
            // Issue next-step loads before consuming current fragments.
            aN.h[0]  = *(const v8h*)&qs[nsub][k + 8 * hh];
            aN.h[1]  = *(const v8h*)&qs[nsub][k + 16 + 8 * hh];
            b0N.h[0] = *(const v8h*)(brow0 + k);
            b0N.h[1] = *(const v8h*)(brow0 + k + 16);
            b1N.h[0] = *(const v8h*)(brow1 + k);
            b1N.h[1] = *(const v8h*)(brow1 + k + 16);
            // Two independent WMMAs sharing A (no D->A/B hazard between them).
            acc0 = wmma_f16(aC.v, b0C.v, acc0);
            acc1 = wmma_f16(aC.v, b1C.v, acc1);
            aC = aN; b0C = b0N; b1C = b1N;
        }
        acc0 = wmma_f16(aC.v, b0C.v, acc0);
        acc1 = wmma_f16(aC.v, b1C.v, acc1);

        // Scatter scores: accX[r] = sim[row = r+8*hh][col tile X, sub-col nsub].
        #pragma unroll
        for (int r = 0; r < 8; ++r) {
            sc[r + 8 * hh][wave * 32 + nsub]      = acc0[r];
            sc[r + 8 * hh][wave * 32 + 16 + nsub] = acc1[r];
        }
        __syncthreads();

        // Running top-4 update: each thread scans its private 16 columns.
        const int gcolBase = colBase + it * COLS_PER_ITER;
        #pragma unroll
        for (int c = 0; c < 16; ++c) {
            const int col = selGrp * 16 + c;
            ins4(sc[selRow][col], gcolBase + col, bestV, bestI);
        }
        __syncthreads();
    }

    // Merge the 16 per-group top-4s of each row.
    #pragma unroll
    for (int t = 0; t < TOPK; ++t) {
        mval[selRow][selGrp * TOPK + t] = bestV[t];
        midx[selRow][selGrp * TOPK + t] = bestI[t];
    }
    __syncthreads();

    if (threadIdx.x < 16) {
        const int row = threadIdx.x;
        float bv[TOPK] = {-2e30f, -2e30f, -2e30f, -2e30f};
        int   bi[TOPK] = {0, 0, 0, 0};
        for (int j = 0; j < 16 * TOPK; ++j) {
            ins4(mval[row][j], midx[row][j], bv, bi);
        }
        const size_t base = ((size_t)(rowBase + row) * NCHUNKS + chunk) * TOPK;
        #pragma unroll
        for (int t = 0; t < TOPK; ++t) {
            cand_val[base + t] = bv[t];
            cand_idx[base + t] = bi[t];
        }
    }
}

// ---------------------------------------------------------------------------
// Kernel 3: per-query merge of chunk candidates -> final top-4 -> gather+mean.
// grid.x = NQ, 256 threads.
// ---------------------------------------------------------------------------
__global__ __launch_bounds__(256) void merge_gather(const float* __restrict__ cand_val,
                                                    const int* __restrict__ cand_idx,
                                                    const float* __restrict__ synth,
                                                    float* __restrict__ out) {
    const int row = blockIdx.x;
    __shared__ int sel[TOPK];
    if (threadIdx.x == 0) {
        float bv[TOPK] = {-2e30f, -2e30f, -2e30f, -2e30f};
        int   bi[TOPK] = {0, 0, 0, 0};
        const float* cv = cand_val + (size_t)row * NCHUNKS * TOPK;
        const int*   ci = cand_idx + (size_t)row * NCHUNKS * TOPK;
        for (int j = 0; j < NCHUNKS * TOPK; ++j) ins4(cv[j], ci[j], bv, bi);
        #pragma unroll
        for (int t = 0; t < TOPK; ++t) sel[t] = bi[t];
    }
    __syncthreads();
    const int i0 = sel[0], i1 = sel[1], i2 = sel[2], i3 = sel[3];
    for (int c = threadIdx.x; c < DDIM; c += 256) {
        float s = synth[(size_t)i0 * DDIM + c] + synth[(size_t)i1 * DDIM + c] +
                  synth[(size_t)i2 * DDIM + c] + synth[(size_t)i3 * DDIM + c];
        out[(size_t)row * DDIM + c] = 0.25f * s;
    }
}

// ---------------------------------------------------------------------------
extern "C" void kernel_launch(void* const* d_in, const int* in_sizes, int n_in,
                              void* d_out, int out_size, void* d_ws, size_t ws_size,
                              hipStream_t stream) {
    const float* query = (const float*)d_in[0];   // [2048, 1024] f32
    const float* match = (const float*)d_in[1];   // [32768, 1024] f32
    const float* synth = (const float*)d_in[2];   // [32768, 1024] f32
    // d_in[3] = topk (fixed at 4 in the reference; compiled in as TOPK)

    char* ws = (char*)d_ws;
    _Float16* Qn = (_Float16*)ws;  ws += (size_t)NQ * DDIM * sizeof(_Float16);   // 4 MB
    _Float16* Mn = (_Float16*)ws;  ws += (size_t)NM * DDIM * sizeof(_Float16);   // 64 MB
    float* cand_val = (float*)ws;  ws += (size_t)NQ * NCHUNKS * TOPK * sizeof(float);
    int*   cand_idx = (int*)ws;    ws += (size_t)NQ * NCHUNKS * TOPK * sizeof(int);

    normalize_f16<<<NQ, 256, 0, stream>>>(query, Qn);
    normalize_f16<<<NM, 256, 0, stream>>>(match, Mn);

    dim3 grid(NCHUNKS, NQ / 16);
    gemm_topk<<<grid, 256, 0, stream>>>(Qn, Mn, cand_val, cand_idx);

    merge_gather<<<NQ, 256, 0, stream>>>(cand_val, cand_idx, synth, (float*)d_out);
}
